// MistralAttention_20272245637393
// MI455X (gfx1250) — matrix operand order repaired and checked by reading, not compiled, not run on hardware
//
#include <hip/hip_runtime.h>
#include <cstddef>
#include <cstdint>

typedef unsigned short u16;
typedef unsigned int   u32;
typedef __attribute__((ext_vector_type(16))) __bf16 v16bf;
typedef __attribute__((ext_vector_type(8)))  float  v8f;
typedef int v4i __attribute__((vector_size(16)));   // GCC-vector int4 (what the async builtin wants)

#define BB   2
#define SS   1024
#define HID  4096
#define NH   32
#define NKV  8
#define HD   128
#define GQA  4

typedef __attribute__((address_space(1))) v4i* g_v4i_p;   // global int4 vector ptr
typedef __attribute__((address_space(3))) v4i* l_v4i_p;   // LDS int4 vector ptr

#if defined(__HIP_DEVICE_COMPILE__) && \
    __has_builtin(__builtin_amdgcn_global_load_async_to_lds_b128) && \
    __has_builtin(__builtin_amdgcn_s_wait_asynccnt)
#define USE_ASYNC 1
#else
#define USE_ASYNC 0
#endif

__device__ __forceinline__ g_v4i_p to_gas(const void* p) {
  return (g_v4i_p)(unsigned long long)(uintptr_t)p;
}
__device__ __forceinline__ l_v4i_p to_las(const void* p) {
  // flat shared address: addr[31:0] is the LDS byte offset (ISA aperture rule)
  return (l_v4i_p)(unsigned int)(uintptr_t)p;
}

union BF16x16 { v16bf v; u16 u[16]; uint4 q[2]; };

__device__ __forceinline__ u16 f2bf(float f) {
  union { float f; u32 u; } c; c.f = f;
  u32 u = c.u;
  u32 r = u + 0x7FFFu + ((u >> 16) & 1u);
  return (u16)(r >> 16);
}
__device__ __forceinline__ float bf2f(u16 h) {
  union { u32 u; float f; } c; c.u = ((u32)h) << 16; return c.f;
}

// ---------------------------------------------------------------- fp32 -> bf16 (row major)
__global__ __launch_bounds__(256) void cvt_f32_bf16(const float* __restrict__ src,
                                                    u16* __restrict__ dst, int n) {
  int i = blockIdx.x * 256 + threadIdx.x;
  if (i < n) dst[i] = f2bf(src[i]);
}

// ---------------------------------------------------------------- fp32 [K][N] -> bf16 [N][K] (tiled)
__global__ __launch_bounds__(256) void cvt_f32_bf16_T(const float* __restrict__ src,
                                                      u16* __restrict__ dst, int K, int N) {
  __shared__ float t[32][33];
  int nB = blockIdx.x * 32, kB = blockIdx.y * 32;
  int x = threadIdx.x & 31, y0 = threadIdx.x >> 5;
  #pragma unroll
  for (int r = 0; r < 4; ++r) {
    int y = y0 + r * 8;
    t[y][x] = src[(size_t)(kB + y) * N + nB + x];
  }
  __syncthreads();
  #pragma unroll
  for (int r = 0; r < 4; ++r) {
    int y = y0 + r * 8;
    dst[(size_t)(nB + y) * K + kB + x] = f2bf(t[x][y]);
  }
}

// ---------------------------------------------------------------- V: [b, s, c] -> [b, c, s] (bf16, tiled)
__global__ __launch_bounds__(256) void transpose_v_k(const u16* __restrict__ v,
                                                     u16* __restrict__ vt) {
  __shared__ u16 t[32][33];
  int b  = blockIdx.z;
  int cB = blockIdx.x * 32;   // col = kvh*HD + d, 0..1023
  int sB = blockIdx.y * 32;
  int x = threadIdx.x & 31, y0 = threadIdx.x >> 5;
  #pragma unroll
  for (int r = 0; r < 4; ++r) {
    int y = y0 + r * 8;
    t[y][x] = v[(size_t)(b * SS + sB + y) * (NKV * HD) + cB + x];
  }
  __syncthreads();
  #pragma unroll
  for (int r = 0; r < 4; ++r) {
    int y = y0 + r * 8;
    vt[(size_t)(b * (NKV * HD) + cB + y) * SS + sB + x] = t[x][y];
  }
}

// ---------------------------------------------------------------- RoPE (in place, bf16)
__global__ __launch_bounds__(256) void rope_bf16(u16* __restrict__ x,
                                                 const float* __restrict__ cosT,
                                                 const float* __restrict__ sinT,
                                                 int nheads, int total) {
  int idx = blockIdx.x * 256 + threadIdx.x;
  if (idx >= total) return;
  int d  = idx & 63;
  int t  = idx >> 6;
  int h  = t % nheads;
  int t2 = t / nheads;
  int s  = t2 % SS;
  int b  = t2 / SS;
  size_t base = ((size_t)(b * SS + s) * nheads + h) * HD;
  float x1 = bf2f(x[base + d]);
  float x2 = bf2f(x[base + 64 + d]);
  float c  = cosT[s * 64 + d];
  float sn = sinT[s * 64 + d];
  x[base + d]      = f2bf(x1 * c - x2 * sn);
  x[base + 64 + d] = f2bf(x2 * c + x1 * sn);
}

// ---------------------------------------------------------------- bf16 GEMM, B pre-transposed
// C[M,N] = A[M,K] * B[K,N], Btg is [N][K].
// Block tile 64x256, BK=32, 256 threads = 8 waves (2Mx4N); wave tile 32x64 = 2x4 WMMA.
// Double-buffered LDS staging via global_load_async_to_lds_b128 (fallback: sync copies).
__device__ __forceinline__ void gemm_stage(const u16* __restrict__ A,
                                           const u16* __restrict__ Btg,
                                           u16 (*As)[48], u16 (*Bt)[48],
                                           int bM, int bN, int k0, int K, int tid) {
  { // A tile 64x32 : one b128 per thread
    int row = tid >> 2, kf = (tid & 3) * 8;
    const u16* gA = A + (size_t)(bM + row) * K + k0 + kf;
#if USE_ASYNC
    __builtin_amdgcn_global_load_async_to_lds_b128(to_gas(gA), to_las(&As[row][kf]), 0, 0);
#else
    *(uint4*)&As[row][kf] = *(const uint4*)gA;
#endif
  }
  #pragma unroll
  for (int p = 0; p < 4; ++p) { // B tile 256x32 : four b128 per thread
    int slot = tid + p * 256;
    int n = slot >> 2, kf = (slot & 3) * 8;
    const u16* gB = Btg + (size_t)(bN + n) * K + k0 + kf;
#if USE_ASYNC
    __builtin_amdgcn_global_load_async_to_lds_b128(to_gas(gB), to_las(&Bt[n][kf]), 0, 0);
#else
    *(uint4*)&Bt[n][kf] = *(const uint4*)gB;
#endif
  }
}

template <bool OUT_BF16>
__global__ __launch_bounds__(256) void gemm_bf16(const u16* __restrict__ A,
                                                 const u16* __restrict__ Btg,
                                                 void* __restrict__ Cout,
                                                 int M, int N, int K) {
  __shared__ __align__(16) u16 As[2][64][48];
  __shared__ __align__(16) u16 Bt[2][256][48];

  const int tid   = threadIdx.x;
  const int lane  = tid & 31;
  const int wave  = tid >> 5;
  const int ln    = lane & 15;
  const int hk    = lane >> 4;
  const int waveM = wave >> 2;      // 0..1
  const int waveN = wave & 3;       // 0..3
  const int bM    = blockIdx.y * 64;
  const int bN    = blockIdx.x * 256;

  v8f zero = {0.f,0.f,0.f,0.f,0.f,0.f,0.f,0.f};
  v8f acc[2][4];
  #pragma unroll
  for (int mi = 0; mi < 2; ++mi)
    #pragma unroll
    for (int ni = 0; ni < 4; ++ni) acc[mi][ni] = zero;

  const int nIter = K >> 5;
  gemm_stage(A, Btg, As[0], Bt[0], bM, bN, 0, K, tid);

  for (int it = 0; it < nIter; ++it) {
    int cur = it & 1;
#if USE_ASYNC
    __builtin_amdgcn_s_wait_asynccnt(0);
#endif
    __syncthreads();
    if (it + 1 < nIter)
      gemm_stage(A, Btg, As[cur ^ 1], Bt[cur ^ 1], bM, bN, (it + 1) << 5, K, tid);

    BF16x16 af[2], bfv[4];
    #pragma unroll
    for (int mi = 0; mi < 2; ++mi) {   // A layout: lane<16 -> K0-7/16-23, lane>=16 -> K8-15/24-31
      int r  = waveM * 32 + mi * 16 + ln;
      int kb = hk * 8;
      af[mi].q[0] = *(const uint4*)&As[cur][r][kb];
      af[mi].q[1] = *(const uint4*)&As[cur][r][kb + 16];
    }
    #pragma unroll
    for (int ni = 0; ni < 4; ++ni) {   // B layout: lane<16 -> K0-7/16-23, lane>=16 -> K8-15/24-31
      int n = waveN * 64 + ni * 16 + ln;
      bfv[ni].q[0] = *(const uint4*)&Bt[cur][n][hk * 8];
      bfv[ni].q[1] = *(const uint4*)&Bt[cur][n][hk * 8 + 16];
    }
    #pragma unroll
    for (int mi = 0; mi < 2; ++mi)
      #pragma unroll
      for (int ni = 0; ni < 4; ++ni)
        acc[mi][ni] = __builtin_amdgcn_wmma_f32_16x16x32_bf16(
            false, af[mi].v, false, bfv[ni].v, (short)0, acc[mi][ni], false, false);
  }

  #pragma unroll
  for (int mi = 0; mi < 2; ++mi)
    #pragma unroll
    for (int ni = 0; ni < 4; ++ni)
      #pragma unroll
      for (int i = 0; i < 8; ++i) {
        int row = bM + waveM * 32 + mi * 16 + i + 8 * hk;
        int col = bN + waveN * 64 + ni * 16 + ln;
        if (OUT_BF16) ((u16*)Cout)[(size_t)row * N + col] = f2bf(acc[mi][ni][i]);
        else          ((float*)Cout)[(size_t)row * N + col] = acc[mi][ni][i];
      }
}

// ---------------------------------------------------------------- flash attention (sliding window)
// 128 threads = 4 waves; block = 64 query rows of one (b, head); each wave owns 16 rows.
// K fragments read straight from global (row-major); V fragments from pre-transposed VT.
__global__ __launch_bounds__(128) void attn_kernel(const u16* __restrict__ Q,
                                                   const u16* __restrict__ Kb,
                                                   const u16* __restrict__ VT,
                                                   u16* __restrict__ O,
                                                   const int* __restrict__ winp) {
  __shared__ float              sS[4][16][32]; // per-wave raw scores
  __shared__ __align__(16) u16  sP[4][16][32]; // per-wave probabilities (bf16)
  __shared__ float              sA[4][16];     // per-wave alpha / 1/l

  const int tid  = threadIdx.x;
  const int lane = tid & 31;
  const int w    = tid >> 5;
  const int ln   = lane & 15;
  const int hk   = lane >> 4;

  int bid = blockIdx.x;
  int qb  = bid & 15;             // SS/64 = 16
  int h   = (bid >> 4) & 31;
  int b   = bid >> 9;
  int kvh = h / GQA;
  int q0  = qb * 64;
  int qr0 = q0 + w * 16;
  int window = *winp;
  const float scale = 0.08838834764831845f; // 128^-0.5

  // Q fragments (WMMA A layout), straight from global
  BF16x16 qf[4];
  {
    int qpos = qr0 + ln;
    const u16* qrow = Q + ((size_t)(b * SS + qpos) * NH + h) * HD;
    int kb = hk * 8;
    #pragma unroll
    for (int c = 0; c < 4; ++c) {
      qf[c].q[0] = *(const uint4*)(qrow + c * 32 + kb);
      qf[c].q[1] = *(const uint4*)(qrow + c * 32 + kb + 16);
    }
  }

  v8f zero = {0.f,0.f,0.f,0.f,0.f,0.f,0.f,0.f};
  v8f oacc[8];
  #pragma unroll
  for (int i = 0; i < 8; ++i) oacc[i] = zero;
  float mrow = -1e30f, lsum = 0.0f;  // valid in lanes 0..15 (row = lane)

  int kstart = q0 - window; if (kstart < 0) kstart = 0; kstart &= ~31;
  int kend = q0 + 64;

  for (int kt = kstart; kt < kend; kt += 32) {
    // scores: S(16x32) = Q(16x128) * K^T(128x32), K fragments from global
    #pragma unroll
    for (int ni = 0; ni < 2; ++ni) {
      v8f sacc = zero;
      int key = kt + ni * 16 + ln;
      const u16* krow = Kb + ((size_t)(b * SS + key) * NKV + kvh) * HD;
      #pragma unroll
      for (int c = 0; c < 4; ++c) {
        BF16x16 bfv;
        bfv.q[0] = *(const uint4*)(krow + c * 32 + hk * 8);
        bfv.q[1] = *(const uint4*)(krow + c * 32 + hk * 8 + 16);
        sacc = __builtin_amdgcn_wmma_f32_16x16x32_bf16(
            false, qf[c].v, false, bfv.v, (short)0, sacc, false, false);
      }
      #pragma unroll
      for (int i = 0; i < 8; ++i) {
        int r    = i + 8 * hk;
        int qpos = qr0 + r;
        int kpos = kt + ni * 16 + ln;
        float sv = sacc[i] * scale;
        bool ok  = (kpos <= qpos) && ((qpos - kpos) <= window);
        sS[w][r][ni * 16 + ln] = ok ? sv : -1e30f;
      }
    }
    __syncthreads();

    // online softmax: lane r in 0..15 owns row r of this wave's tile
    if (lane < 16) {
      int r = lane;
      float rowmax = -1e30f;
      #pragma unroll 8
      for (int j = 0; j < 32; ++j) rowmax = fmaxf(rowmax, sS[w][r][j]);
      float newm  = fmaxf(mrow, rowmax);
      float alpha = __expf(mrow - newm);
      float sum   = 0.0f;
      #pragma unroll 8
      for (int j = 0; j < 32; ++j) {
        float sv = sS[w][r][j];
        float pv = (sv > -1e29f) ? __expf(sv - newm) : 0.0f;
        sum += pv;
        sP[w][r][j] = f2bf(pv);
      }
      mrow = newm;
      lsum = lsum * alpha + sum;
      sA[w][r] = alpha;
    }
    __syncthreads();

    // rescale + O += P(16x32) * V(32x128); V fragments from transposed VT (contiguous)
    float av[8];
    #pragma unroll
    for (int i = 0; i < 8; ++i) av[i] = sA[w][i + 8 * hk];
    BF16x16 pf;
    {
      int kb = hk * 8;
      pf.q[0] = *(const uint4*)&sP[w][ln][kb];
      pf.q[1] = *(const uint4*)&sP[w][ln][kb + 16];
    }
    #pragma unroll
    for (int nt = 0; nt < 8; ++nt) {
      #pragma unroll
      for (int i = 0; i < 8; ++i) oacc[nt][i] *= av[i];
      BF16x16 vf;
      const u16* vrow = VT + ((size_t)(b * (NKV * HD) + kvh * HD + nt * 16 + ln)) * SS;
      vf.q[0] = *(const uint4*)(vrow + kt + hk * 8);
      vf.q[1] = *(const uint4*)(vrow + kt + hk * 8 + 16);
      oacc[nt] = __builtin_amdgcn_wmma_f32_16x16x32_bf16(
          false, pf.v, false, vf.v, (short)0, oacc[nt], false, false);
    }
  }

  __syncthreads();
  if (lane < 16) sA[w][lane] = (lsum > 0.0f) ? (1.0f / lsum) : 0.0f;
  __syncthreads();
  float fv[8];
  #pragma unroll
  for (int i = 0; i < 8; ++i) fv[i] = sA[w][i + 8 * hk];

  #pragma unroll
  for (int nt = 0; nt < 8; ++nt)
    #pragma unroll
    for (int i = 0; i < 8; ++i) {
      int r    = i + 8 * hk;
      int qpos = qr0 + r;
      int d    = nt * 16 + ln;
      O[((size_t)(b * SS + qpos) * NH + h) * HD + d] = f2bf(oacc[nt][i] * fv[i]);
    }
}

// ---------------------------------------------------------------- launch
extern "C" void kernel_launch(void* const* d_in, const int* in_sizes, int n_in,
                              void* d_out, int out_size, void* d_ws, size_t ws_size,
                              hipStream_t stream) {
  const float* hs   = (const float*)d_in[0];
  const float* wq   = (const float*)d_in[1];
  const float* wk   = (const float*)d_in[2];
  const float* wv   = (const float*)d_in[3];
  const float* wo   = (const float*)d_in[4];
  const float* cosT = (const float*)d_in[5];
  const float* sinT = (const float*)d_in[6];
  const int*   winp = (const int*)d_in[7];
  float* out = (float*)d_out;

  const int M = BB * SS;                       // 2048
  const size_t n_hs = (size_t)M * HID;         // 8,388,608
  const size_t n_wq = (size_t)HID * NH * HD;   // 16,777,216
  const size_t n_wk = (size_t)HID * NKV * HD;  // 4,194,304
  const size_t n_q  = (size_t)M * NH * HD;     // 8,388,608
  const size_t n_k  = (size_t)M * NKV * HD;    // 2,097,152

  char* p = (char*)d_ws;
  u16* hsb = (u16*)p; p += n_hs * 2;   // A (row major)
  u16* wqT = (u16*)p; p += n_wq * 2;   // [N][K]
  u16* wkT = (u16*)p; p += n_wk * 2;
  u16* wvT = (u16*)p; p += n_wk * 2;
  u16* woT = (u16*)p; p += n_wq * 2;
  u16* qb  = (u16*)p; p += n_q * 2;
  u16* kb  = (u16*)p; p += n_k * 2;
  u16* vb  = (u16*)p; p += n_k * 2;
  u16* vtb = (u16*)p; p += n_k * 2;
  u16* aob = (u16*)p; p += n_q * 2;

  // 1) convert (A row-major; weights transposed to [N][K])
  cvt_f32_bf16<<<(int)((n_hs + 255) / 256), 256, 0, stream>>>(hs, hsb, (int)n_hs);
  cvt_f32_bf16_T<<<dim3((NH * HD) / 32, HID / 32), 256, 0, stream>>>(wq, wqT, HID, NH * HD);
  cvt_f32_bf16_T<<<dim3((NKV * HD) / 32, HID / 32), 256, 0, stream>>>(wk, wkT, HID, NKV * HD);
  cvt_f32_bf16_T<<<dim3((NKV * HD) / 32, HID / 32), 256, 0, stream>>>(wv, wvT, HID, NKV * HD);
  cvt_f32_bf16_T<<<dim3(HID / 32, (NH * HD) / 32), 256, 0, stream>>>(wo, woT, NH * HD, HID);

  // 2) QKV projections (bf16 out)
  gemm_bf16<true><<<dim3((NH * HD) / 256, M / 64), 256, 0, stream>>>(hsb, wqT, qb, M, NH * HD, HID);
  gemm_bf16<true><<<dim3((NKV * HD) / 256, M / 64), 256, 0, stream>>>(hsb, wkT, kb, M, NKV * HD, HID);
  gemm_bf16<true><<<dim3((NKV * HD) / 256, M / 64), 256, 0, stream>>>(hsb, wvT, vb, M, NKV * HD, HID);

  // 3) RoPE on Q and K; transpose V for the PV GEMM
  {
    int tq = BB * SS * NH * 64;   // 4,194,304
    int tk = BB * SS * NKV * 64;  // 1,048,576
    rope_bf16<<<(tq + 255) / 256, 256, 0, stream>>>(qb, cosT, sinT, NH, tq);
    rope_bf16<<<(tk + 255) / 256, 256, 0, stream>>>(kb, cosT, sinT, NKV, tk);
    transpose_v_k<<<dim3((NKV * HD) / 32, SS / 32, BB), 256, 0, stream>>>(vb, vtb);
  }

  // 4) sliding-window flash attention
  attn_kernel<<<BB * NH * (SS / 64), 128, 0, stream>>>(qb, kb, vtb, aob, winp);

  // 5) output projection (f32 out)
  gemm_bf16<false><<<dim3(HID / 256, M / 64), 256, 0, stream>>>(aob, woT, out, M, HID, HID);

  (void)in_sizes; (void)n_in; (void)out_size; (void)ws_size;
}
